// PatchSSMClassifier_44667659878986
// MI455X (gfx1250) — compile-verified, matrix operand order checked by reading, not run on hardware
//
#include <hip/hip_runtime.h>
#include <hip/hip_bf16.h>
#include <math.h>

// ---------------------------------------------------------------------------
// PatchSSMClassifier for MI455X (gfx1250, wave32, WMMA)
//   patches@WpT -> scan_{t=0..255}{ s=s@A^T+u_t@Bm^T ; y_t=s@Cm^T }
//   -> gelu -> LN -> mean_L -> @Wf^T
// All GEMMs on v_wmma_f32_16x16x32_bf16 (f32 accumulate). Weights/activations
// converted to bf16 once; everything L2-resident (A+Bm+Cm bf16 = 4MB).
// ---------------------------------------------------------------------------

typedef __bf16 bf16_t;
typedef __attribute__((ext_vector_type(16))) __bf16 v16bf;
typedef __attribute__((ext_vector_type(8)))  __bf16 v8bf;
typedef __attribute__((ext_vector_type(8)))  float  v8f;

#define BATCH    128
#define SEQ_LEN  256
#define D_MODEL  512
#define D_STATE  1024
#define N_CLS    1000

union FragU { v16bf v; v8bf h[2]; };

// 16x32 bf16 operand fragment, ISA 7.12.2 "16-bit A-Matrix 16x32" layout:
// lanes 0-15 : row = row0+lane,    elems 0-7 -> K=k0..k0+7,  elems 8-15 -> K=k0+16..k0+23
// lanes 16-31: row = row0+lane-16, elems 0-7 -> K=k0+8..+15, elems 8-15 -> K=k0+24..+31
// => per lane: two contiguous 16B chunks of a row-major [row][K] matrix.
// Used for both A (activation rows) and B (weight rows, computing X @ W^T).
__device__ __forceinline__ v16bf load_frag(const bf16_t* __restrict__ base,
                                           int row0, int k0, int ld) {
  const int lane = threadIdx.x & 31;
  const int r  = row0 + (lane & 15);
  const int kb = k0 + ((lane & 16) >> 1);          // +8 for upper half-wave
  const bf16_t* p = base + (size_t)r * ld + kb;
  FragU f;
  f.h[0] = *(const v8bf*)(p);                      // K = kb   .. kb+7
  f.h[1] = *(const v8bf*)(p + 16);                 // K = kb+16.. kb+23
  return f.v;
}

__device__ __forceinline__ v8f wmma_bf16(v16bf a, v16bf b, v8f c) {
  return __builtin_amdgcn_wmma_f32_16x16x32_bf16(false, a, false, b,
                                                 (short)0, c, false, false);
}

// 16x16 f32 C/D layout: lane 0-15 -> N=lane, M=vgpr; lane 16-31 -> N=lane-16, M=vgpr+8
__device__ __forceinline__ void store_tile_f32(float* __restrict__ dst, int m0, int n0,
                                               int ld, v8f acc) {
  const int lane = threadIdx.x & 31;
  const int n  = n0 + (lane & 15);
  const int mb = m0 + ((lane & 16) >> 1);
#pragma unroll
  for (int v = 0; v < 8; ++v) dst[(size_t)(mb + v) * ld + n] = acc[v];
}

__device__ __forceinline__ void store_tile_bf16(bf16_t* __restrict__ dst, int m0, int n0,
                                                int ld, v8f acc) {
  const int lane = threadIdx.x & 31;
  const int n  = n0 + (lane & 15);
  const int mb = m0 + ((lane & 16) >> 1);
#pragma unroll
  for (int v = 0; v < 8; ++v) dst[(size_t)(mb + v) * ld + n] = (bf16_t)acc[v];
}

// ---------------------------------------------------------------------------
// f32 -> bf16 conversion
__global__ __launch_bounds__(256)
void cvt_bf16_kernel(const float* __restrict__ src, bf16_t* __restrict__ dst, int n) {
  int i = blockIdx.x * 256 + threadIdx.x;
  if (i < n) dst[i] = (bf16_t)src[i];
}

// ---------------------------------------------------------------------------
// Patch embed: x[l][b][d] = patch(b,l) . Wp[d] + bp[d]   (bf16 out)
__global__ __launch_bounds__(64)
void patch_embed_kernel(const float* __restrict__ images,   // [128][3][64][64]
                        const float* __restrict__ Wp,       // [512][48]
                        const float* __restrict__ bp,       // [512]
                        bf16_t* __restrict__ x)             // [256][128][512]
{
  __shared__ float patch[48];
  const int bl = blockIdx.x;                 // b*256 + l
  const int b  = bl >> 8, l = bl & 255;
  const int ph = l >> 4, pw = l & 15;
  const int tid = threadIdx.x;
  if (tid < 48) {
    const int c = tid / 16, r = tid & 15;
    const int py = r >> 2, px = r & 3;       // patch elem index = c*16 + py*4 + px
    patch[tid] = images[(((size_t)b * 3 + c) * 64 + (ph * 4 + py)) * 64 + (pw * 4 + px)];
  }
  __syncthreads();
  for (int d = tid; d < D_MODEL; d += 64) {
    float acc = bp[d];
    const float* w = Wp + (size_t)d * 48;
#pragma unroll
    for (int k = 0; k < 48; ++k) acc += patch[k] * w[k];
    x[((size_t)l * BATCH + b) * D_MODEL + d] = (bf16_t)acc;
  }
}

// ---------------------------------------------------------------------------
// One scan step.  256 threads = 8 waves; each wave owns a 32x32 macro-tile
// (2x2 WMMA tiles).  Blocks [0,16): state_out = state_in@A^T + u@Bm^T (N=1024).
// Blocks [16,24) (or [0,8) when do_update==0): y_out = state_in@Cm^T (N=512).
__global__ __launch_bounds__(256)
void ssm_step_kernel(const bf16_t* __restrict__ state_in,   // [128][1024]
                     const bf16_t* __restrict__ A,          // [1024][1024]
                     const bf16_t* __restrict__ Bm,         // [1024][512]
                     const bf16_t* __restrict__ Cm,         // [512][1024]
                     const bf16_t* __restrict__ u,          // [128][512]
                     bf16_t* __restrict__ state_out,        // [128][1024]
                     float*  __restrict__ y_out,            // [128][512] or null
                     int do_update)
{
  const int wave = threadIdx.x >> 5;
  const int bid  = blockIdx.x;
  const int m0   = (wave >> 1) * 32;                        // batch rows
  v8f acc00 = {}, acc01 = {}, acc10 = {}, acc11 = {};

  if (do_update && bid < 16) {
    const int n0 = bid * 64 + (wave & 1) * 32;              // state cols
    for (int k0 = 0; k0 < D_STATE; k0 += 32) {              // state_in @ A^T
      v16bf a0 = load_frag(state_in, m0,      k0, D_STATE);
      v16bf a1 = load_frag(state_in, m0 + 16, k0, D_STATE);
      v16bf b0 = load_frag(A,        n0,      k0, D_STATE);
      v16bf b1 = load_frag(A,        n0 + 16, k0, D_STATE);
      acc00 = wmma_bf16(a0, b0, acc00);
      acc01 = wmma_bf16(a0, b1, acc01);
      acc10 = wmma_bf16(a1, b0, acc10);
      acc11 = wmma_bf16(a1, b1, acc11);
    }
    for (int k0 = 0; k0 < D_MODEL; k0 += 32) {              // + u @ Bm^T
      v16bf a0 = load_frag(u,  m0,      k0, D_MODEL);
      v16bf a1 = load_frag(u,  m0 + 16, k0, D_MODEL);
      v16bf b0 = load_frag(Bm, n0,      k0, D_MODEL);
      v16bf b1 = load_frag(Bm, n0 + 16, k0, D_MODEL);
      acc00 = wmma_bf16(a0, b0, acc00);
      acc01 = wmma_bf16(a0, b1, acc01);
      acc10 = wmma_bf16(a1, b0, acc10);
      acc11 = wmma_bf16(a1, b1, acc11);
    }
    store_tile_bf16(state_out, m0,      n0,      D_STATE, acc00);
    store_tile_bf16(state_out, m0,      n0 + 16, D_STATE, acc01);
    store_tile_bf16(state_out, m0 + 16, n0,      D_STATE, acc10);
    store_tile_bf16(state_out, m0 + 16, n0 + 16, D_STATE, acc11);
  } else {
    const int yb = do_update ? (bid - 16) : bid;
    const int n0 = yb * 64 + (wave & 1) * 32;               // y cols
    for (int k0 = 0; k0 < D_STATE; k0 += 32) {              // state_in @ Cm^T
      v16bf a0 = load_frag(state_in, m0,      k0, D_STATE);
      v16bf a1 = load_frag(state_in, m0 + 16, k0, D_STATE);
      v16bf b0 = load_frag(Cm,       n0,      k0, D_STATE);
      v16bf b1 = load_frag(Cm,       n0 + 16, k0, D_STATE);
      acc00 = wmma_bf16(a0, b0, acc00);
      acc01 = wmma_bf16(a0, b1, acc01);
      acc10 = wmma_bf16(a1, b0, acc10);
      acc11 = wmma_bf16(a1, b1, acc11);
    }
    store_tile_f32(y_out, m0,      n0,      D_MODEL, acc00);
    store_tile_f32(y_out, m0,      n0 + 16, D_MODEL, acc01);
    store_tile_f32(y_out, m0 + 16, n0,      D_MODEL, acc10);
    store_tile_f32(y_out, m0 + 16, n0 + 16, D_MODEL, acc11);
  }
}

// ---------------------------------------------------------------------------
// gelu(exact) -> layernorm over D -> accumulate mean over L into pooled
__device__ __forceinline__ float gelu_f(float x) {
  return 0.5f * x * (1.0f + erff(x * 0.70710678118654752f));
}

__global__ __launch_bounds__(256)
void post_kernel(const float* __restrict__ y,       // [128][512] one step
                 const float* __restrict__ gamma,
                 const float* __restrict__ beta,
                 float* __restrict__ pooled)        // [128][512] running sum
{
  __shared__ float ssum[8], ssq[8];
  const int b = blockIdx.x, tid = threadIdx.x;
  const float* yr = y + (size_t)b * D_MODEL;
  float g0 = gelu_f(yr[tid]);
  float g1 = gelu_f(yr[tid + 256]);
  float s = g0 + g1;
  float q = g0 * g0 + g1 * g1;
#pragma unroll
  for (int o = 16; o > 0; o >>= 1) {                // wave32 reduction
    s += __shfl_xor(s, o, 32);
    q += __shfl_xor(q, o, 32);
  }
  const int wv = tid >> 5;
  if ((tid & 31) == 0) { ssum[wv] = s; ssq[wv] = q; }
  __syncthreads();
  if (tid == 0) {
    float S = 0.f, Q = 0.f;
#pragma unroll
    for (int i = 0; i < 8; ++i) { S += ssum[i]; Q += ssq[i]; }
    ssum[0] = S; ssq[0] = Q;
  }
  __syncthreads();
  const float mean = ssum[0] * (1.0f / (float)D_MODEL);
  const float var  = ssq[0]  * (1.0f / (float)D_MODEL) - mean * mean;
  const float inv  = rsqrtf(var + 1e-5f);
  const float z0 = (g0 - mean) * inv * gamma[tid]       + beta[tid];
  const float z1 = (g1 - mean) * inv * gamma[tid + 256] + beta[tid + 256];
  pooled[(size_t)b * D_MODEL + tid]       += z0 * (1.0f / (float)SEQ_LEN);
  pooled[(size_t)b * D_MODEL + tid + 256] += z1 * (1.0f / (float)SEQ_LEN);
}

// ---------------------------------------------------------------------------
// logits[b][c] = pooled[b] . Wf[c] + bf[c]
__global__ __launch_bounds__(128)
void logits_kernel(const float* __restrict__ pooled,  // [128][512]
                   const float* __restrict__ Wf,      // [1000][512]
                   const float* __restrict__ bfv,     // [1000]
                   float* __restrict__ out)           // [128][1000]
{
  __shared__ float pr[D_MODEL];
  const int b = blockIdx.x;
  for (int i = threadIdx.x; i < D_MODEL; i += 128)
    pr[i] = pooled[(size_t)b * D_MODEL + i];
  __syncthreads();
  const int c = blockIdx.y * 128 + threadIdx.x;
  if (c < N_CLS) {
    const float* w = Wf + (size_t)c * D_MODEL;
    float acc = bfv[c];
    for (int k = 0; k < D_MODEL; ++k) acc += pr[k] * w[k];
    out[(size_t)b * N_CLS + c] = acc;
  }
}

// ---------------------------------------------------------------------------
extern "C" void kernel_launch(void* const* d_in, const int* in_sizes, int n_in,
                              void* d_out, int out_size, void* d_ws, size_t ws_size,
                              hipStream_t stream) {
  const float* images = (const float*)d_in[0];
  const float* Wp     = (const float*)d_in[1];
  const float* bp     = (const float*)d_in[2];
  const float* A      = (const float*)d_in[3];
  const float* Bm     = (const float*)d_in[4];
  const float* Cm     = (const float*)d_in[5];
  const float* gamma  = (const float*)d_in[6];
  const float* beta   = (const float*)d_in[7];
  const float* Wf     = (const float*)d_in[8];
  const float* bfv    = (const float*)d_in[9];
  float* out = (float*)d_out;

  char* ws = (char*)d_ws;
  size_t off = 0;
  auto take = [&](size_t bytes) -> void* {
    off = (off + 255) & ~(size_t)255;
    void* p = ws + off;
    off += bytes;
    return p;
  };

  bf16_t* A_bf  = (bf16_t*)take((size_t)D_STATE * D_STATE * 2);
  bf16_t* Bm_bf = (bf16_t*)take((size_t)D_STATE * D_MODEL * 2);
  bf16_t* Cm_bf = (bf16_t*)take((size_t)D_MODEL * D_STATE * 2);
  bf16_t* x_bf  = (bf16_t*)take((size_t)SEQ_LEN * BATCH * D_MODEL * 2);
  bf16_t* st0   = (bf16_t*)take((size_t)BATCH * D_STATE * 2);
  bf16_t* st1   = (bf16_t*)take((size_t)BATCH * D_STATE * 2);
  float*  ystep = (float*)take((size_t)BATCH * D_MODEL * 4);
  float*  pooled= (float*)take((size_t)BATCH * D_MODEL * 4);

  // one-time bf16 conversion of the scan weights
  cvt_bf16_kernel<<<(D_STATE * D_STATE + 255) / 256, 256, 0, stream>>>(A, A_bf, D_STATE * D_STATE);
  cvt_bf16_kernel<<<(D_STATE * D_MODEL + 255) / 256, 256, 0, stream>>>(Bm, Bm_bf, D_STATE * D_MODEL);
  cvt_bf16_kernel<<<(D_MODEL * D_STATE + 255) / 256, 256, 0, stream>>>(Cm, Cm_bf, D_MODEL * D_STATE);

  hipMemsetAsync(st0, 0, (size_t)BATCH * D_STATE * 2, stream);
  hipMemsetAsync(pooled, 0, (size_t)BATCH * D_MODEL * 4, stream);

  patch_embed_kernel<<<BATCH * SEQ_LEN, 64, 0, stream>>>(images, Wp, bp, x_bf);

  bf16_t* cur = st0;
  bf16_t* nxt = st1;
  for (int t = 0; t < SEQ_LEN; ++t) {
    const int grid = (t > 0) ? 24 : 16;      // 16 state-update WGs + 8 y WGs
    ssm_step_kernel<<<grid, 256, 0, stream>>>(cur, A_bf, Bm_bf, Cm_bf,
                                              x_bf + (size_t)t * BATCH * D_MODEL,
                                              nxt, (t > 0) ? ystep : nullptr, 1);
    if (t > 0)
      post_kernel<<<BATCH, 256, 0, stream>>>(ystep, gamma, beta, pooled);
    bf16_t* tmp = cur; cur = nxt; nxt = tmp;
  }
  // final y_{L-1} = s_L @ Cm^T
  ssm_step_kernel<<<8, 256, 0, stream>>>(cur, A_bf, Bm_bf, Cm_bf, x_bf,
                                         nullptr, ystep, 0);
  post_kernel<<<BATCH, 256, 0, stream>>>(ystep, gamma, beta, pooled);

  logits_kernel<<<dim3(BATCH, 8), 128, 0, stream>>>(pooled, Wf, bfv, out);
}
